// WindowAttention_62311385531151
// MI455X (gfx1250) — matrix operand order repaired and checked by reading, not compiled, not run on hardware
//
#include <hip/hip_runtime.h>
#include <hip/hip_bf16.h>
#include <stdint.h>

typedef _Float16 h8   __attribute__((ext_vector_type(8)));
typedef _Float16 v16h __attribute__((ext_vector_type(16)));
typedef float    v8f  __attribute__((ext_vector_type(8)));
typedef unsigned int tdm_v4u __attribute__((ext_vector_type(4)));
typedef int          tdm_v8i __attribute__((ext_vector_type(8)));
typedef int          tdm_v4i __attribute__((ext_vector_type(4)));

// ---------------------------------------------------------------------------
// WMMA fragment builders (wave32, v_wmma_f32_16x16x32_f16 layouts).
// A (16x32, MxK): lane m = lane%16; halves 0..7 hold K = kb..kb+7,
//                 halves 8..15 hold K = kb+16..kb+23, kb = (lane<16)?0:8.
// B (32x16, KxN): lane n = lane%16; halves 0..7 hold K = kb..kb+7,
//                 halves 8..15 hold K = kb+16..kb+23, kb = (lane<16)?0:8.
// ---------------------------------------------------------------------------
__device__ __forceinline__ v16h frag_a(const _Float16* p, int lane) {
    int kb = (lane & 16) ? 8 : 0;
    union { v16h v; h8 h[2]; } u;
    u.h[0] = *reinterpret_cast<const h8*>(p + kb);
    u.h[1] = *reinterpret_cast<const h8*>(p + kb + 16);
    return u.v;
}
__device__ __forceinline__ v16h frag_b(const _Float16* p, int lane) {
    int kb = (lane & 16) ? 8 : 0;
    union { v16h v; h8 h[2]; } u;
    u.h[0] = *reinterpret_cast<const h8*>(p + kb);
    u.h[1] = *reinterpret_cast<const h8*>(p + kb + 16);
    return u.v;
}
__device__ __forceinline__ v8f wmma_f16(v16h a, v16h b, v8f c) {
    return __builtin_amdgcn_wmma_f32_16x16x32_f16(false, a, false, b, (short)0, c,
                                                  false, false);
}

// ---------------------------------------------------------------------------
// TDM: load a 2D f16 tile (tile0 contiguous elems per row, tile1 rows,
// row stride = stride0 elems) from global into LDS, packed contiguously.
// D# packing per CDNA5 ISA (count=1, data_size=2B, type=2).
// This toolchain uses the 6-arg builtin: (g0 u32x4, g1 i32x8, g2 i32x4,
// g3 i32x4, g4 i32x8, cpol i32) — trailing groups zero for a 2D tile.
// ---------------------------------------------------------------------------
__device__ __forceinline__ void tdm_load_2d_f16(const _Float16* gptr,
                                                unsigned lds_off,
                                                unsigned tile0, unsigned tile1,
                                                unsigned stride0) {
    unsigned long long ga = (unsigned long long)(uintptr_t)gptr;
    tdm_v4u g0;
    g0[0] = 1u;                                        // count=1, user mode
    g0[1] = lds_off;                                   // lds_addr (bytes)
    g0[2] = (unsigned)(ga & 0xffffffffu);              // global_addr[31:0]
    g0[3] = (unsigned)((ga >> 32) & 0x01ffffffu)       // global_addr[56:32]
            | (2u << 30);                              // type=2 (image)
    tdm_v8i g1;
    g1[0] = 0x00010000;                                // data_size=1 (2 bytes)
    g1[1] = (int)((tile0 & 0xffffu) << 16);            // tensor_dim0[15:0]
    g1[2] = (int)((tile1 & 0xffffu) << 16);            // tdim0 hi=0 | tensor_dim1 lo
    g1[3] = (int)((tile0 & 0xffffu) << 16);            // tdim1 hi=0 | tile_dim0
    g1[4] = (int)(tile1 & 0xffffu);                    // tile_dim1 | tile_dim2=0
    g1[5] = (int)stride0;                              // tensor_dim0_stride[31:0]
    g1[6] = 0;                                         // stride0 hi | dim1_stride lo
    g1[7] = 0;
    tdm_v4i z4 = {0, 0, 0, 0};
    tdm_v8i z8 = {0, 0, 0, 0, 0, 0, 0, 0};
    __builtin_amdgcn_tensor_load_to_lds(g0, g1, z4, z4, z8, 0);
}
__device__ __forceinline__ unsigned lds_off32(const void* p) {
    return (unsigned)(uintptr_t)p;                     // low 32b = LDS byte offset
}

// ---------------------------------------------------------------------------
// Kernel A: convert + TRANSPOSE weights to f16: wqT[n][k] 768x256,
// woT[n][k] 256x256 (so GEMM B-tiles are contiguous-row 2D blocks for TDM)
// ---------------------------------------------------------------------------
__global__ void wa_cvt_kernel(const float* __restrict__ wqkv,
                              const float* __restrict__ wout,
                              _Float16* __restrict__ wqT,
                              _Float16* __restrict__ woT) {
    int i = blockIdx.x * 256 + threadIdx.x;            // grid covers 768*256
    if (i < 768 * 256) {
        int n = i >> 8, kd = i & 255;
        wqT[i] = (_Float16)wqkv[kd * 768 + n];
    }
    if (i < 256 * 256) {
        int n = i >> 8, kd = i & 255;
        woT[i] = (_Float16)wout[kd * 256 + n];
    }
}

// ---------------------------------------------------------------------------
// Kernel C: continuous-position-bias MLP -> bias[head][i][j]  (8 x 64 x 64)
// ---------------------------------------------------------------------------
__global__ void wa_cpb_kernel(const float* __restrict__ w1,   // (2,512)
                              const float* __restrict__ b1,   // (512)
                              const float* __restrict__ w2,   // (512,8)
                              float* __restrict__ bias) {
    int p = blockIdx.x * 64 + threadIdx.x;   // 0..4095 = i*64+j
    int i = p >> 6, j = p & 63;
    int yi = i >> 3, xi = i & 7, yj = j >> 3, xj = j & 7;
    float dy = (float)(yi - yj), dx = (float)(xi - xj);
    const float inv3 = 1.0f / 3.0f;          // 1/log2(8)
    float sy = (dy > 0.f) ? 1.f : ((dy < 0.f) ? -1.f : 0.f);
    float sx = (dx > 0.f) ? 1.f : ((dx < 0.f) ? -1.f : 0.f);
    float ry = sy * log2f(1.f + fabsf(dy)) * inv3;
    float rx = sx * log2f(1.f + fabsf(dx)) * inv3;
    float out[8] = {0.f, 0.f, 0.f, 0.f, 0.f, 0.f, 0.f, 0.f};
    for (int kk = 0; kk < 512; ++kk) {
        float h = fmaxf(ry * w1[kk] + rx * w1[512 + kk] + b1[kk], 0.f);
        #pragma unroll
        for (int e = 0; e < 8; ++e) out[e] += h * w2[kk * 8 + e];
    }
    #pragma unroll
    for (int e = 0; e < 8; ++e) bias[e * 4096 + p] = out[e];
}

// ---------------------------------------------------------------------------
// Kernel S: materialize cyclically shifted input channel-last as f16:
// xh[b][y'][x'][c] = (f16) x[b][c][(y'+4)&63][(x'+4)&63]
// Coalesced reads along x; LDS bounce; 16B coalesced stores.
// ---------------------------------------------------------------------------
__global__ __launch_bounds__(256) void wa_shift_kernel(const float* __restrict__ x,
                                                       _Float16* __restrict__ xh) {
    __shared__ __align__(16) _Float16 tile[64 * 256];  // [px][c]
    int blk = blockIdx.x;                              // b*64 + y'
    int b = blk >> 6, yrow = blk & 63;
    int ysrc = (yrow + 4) & 63;
    int t = threadIdx.x;
    for (int i = t; i < 16384; i += 256) {
        int c = i >> 6, px = i & 63;
        float v = x[(((size_t)b * 256 + c) * 64 + ysrc) * 64 + ((px + 4) & 63)];
        tile[px * 256 + c] = (_Float16)v;
    }
    __syncthreads();
    size_t base = (size_t)blk * 64 * 256;
    for (int i = t; i < 2048; i += 256) {
        *reinterpret_cast<h8*>(xh + base + (size_t)i * 8) =
            *reinterpret_cast<const h8*>(tile + i * 8);
    }
}

// ---------------------------------------------------------------------------
// Kernel B: QKV GEMM (M=131072 x K=256 x N=768). A/B tiles DMA'd into LDS by
// the Tensor Data Mover (one 16x32 f16 tile per wave per operand per K-step).
// Scatter into windowed layouts:
//   q,k : [b][head][win][token(64)][dim(32)]  (f32)
//   v   : [b][head][win][dim(32)][token(64)]  (f32, transposed)
// ---------------------------------------------------------------------------
__global__ __launch_bounds__(128) void wa_qkv_gemm(
        const _Float16* __restrict__ xh, const _Float16* __restrict__ wqT,
        float* __restrict__ q, float* __restrict__ k, float* __restrict__ vt) {
    __shared__ __align__(16) _Float16 aT[64 * 32];     // [px][k]
    __shared__ __align__(16) _Float16 bT[64 * 32];     // [n][k]
    int mt = blockIdx.x / 12, nt = blockIdx.x % 12;
    int b = mt >> 6, yrow = mt & 63;                   // y' (post-shift row)
    int t = threadIdx.x, lane = t & 31, wid = t >> 5;
    const _Float16* aSrc = xh + (size_t)mt * 64 * 256 + (size_t)(wid * 16) * 256;
    const _Float16* bSrc = wqT + ((size_t)nt * 64 + wid * 16) * 256;
    unsigned aOff = lds_off32(&aT[wid * 16 * 32]);
    unsigned bOff = lds_off32(&bT[wid * 16 * 32]);
    v8f acc[4] = {};
    for (int kk = 0; kk < 8; ++kk) {
        tdm_load_2d_f16(aSrc + kk * 32, aOff, 32, 16, 256);   // wave's A quarter
        tdm_load_2d_f16(bSrc + kk * 32, bOff, 32, 16, 256);   // wave's B quarter
        __builtin_amdgcn_s_wait_tensorcnt(0);
        __syncthreads();
        int row = wid * 16 + (lane & 15);
        v16h a = frag_a(&aT[row * 32], lane);
        #pragma unroll
        for (int nj = 0; nj < 4; ++nj) {
            v16h bf = frag_b(&bT[(nj * 16 + (lane & 15)) * 32], lane);
            acc[nj] = wmma_f16(a, bf, acc[nj]);
        }
        __syncthreads();
    }
    int wy = yrow >> 3, ty = yrow & 7;
    int mh = ((lane >> 4) & 1) * 8;
    #pragma unroll
    for (int nj = 0; nj < 4; ++nj) {
        int n = nt * 64 + nj * 16 + (lane & 15);       // 0..767
        int which = n >> 8;                            // 0=q,1=k,2=v (uniform)
        int inner = n & 255;
        int head = inner >> 5, d = inner & 31;
        #pragma unroll
        for (int r = 0; r < 8; ++r) {
            int px = wid * 16 + r + mh;
            int wx = px >> 3, tx = px & 7;
            int win = wy * 8 + wx, tl = ty * 8 + tx;
            size_t base = ((size_t)b * 8 + head) * 64 + win;
            float val = acc[nj][r];
            if (which == 0)      q[(base * 64 + tl) * 32 + d] = val;
            else if (which == 1) k[(base * 64 + tl) * 32 + d] = val;
            else                 vt[(base * 32 + d) * 64 + tl] = val;
        }
    }
}

// ---------------------------------------------------------------------------
// Kernel E: windowed cosine attention. 1 wave per (b, head, window).
// LDS: qh/kh 64x32 f16, vh = V^T 32x64 f16, sc 64x64 f32, ah 64x64 f16 (36KB)
// Output written as f16 (feeds the f16 WMMA output GEMM via TDM).
// ---------------------------------------------------------------------------
__global__ __launch_bounds__(32) void wa_attn_kernel(
        const float* __restrict__ q, const float* __restrict__ k,
        const float* __restrict__ vt, const float* __restrict__ tau,
        const float* __restrict__ bias, _Float16* __restrict__ ao) {
    __shared__ __align__(16) _Float16 qh[64 * 32];
    __shared__ __align__(16) _Float16 kh[64 * 32];
    __shared__ __align__(16) _Float16 vh[32 * 64];
    __shared__ __align__(16) float    sc[64 * 64];
    __shared__ __align__(16) _Float16 ah[64 * 64];
    int idx = blockIdx.x;
    int b = idx >> 9, head = (idx >> 6) & 7, win = idx & 63;
    int lane = threadIdx.x;
    size_t base = ((size_t)b * 8 + head) * 64 + win;
    const float* qg = q + base * 64 * 32;
    const float* kg = k + base * 64 * 32;
    const float* vg = vt + base * 32 * 64;

    #pragma unroll
    for (int rr = 0; rr < 2; ++rr) {                   // cosine normalize q,k
        int row = lane * 2 + rr;
        float s = 0.f;
        for (int d = 0; d < 32; ++d) { float v = qg[row * 32 + d]; s += v * v; }
        float inv = rsqrtf(fmaxf(s, 1e-24f));
        for (int d = 0; d < 32; ++d) qh[row * 32 + d] = (_Float16)(qg[row * 32 + d] * inv);
        s = 0.f;
        for (int d = 0; d < 32; ++d) { float v = kg[row * 32 + d]; s += v * v; }
        inv = rsqrtf(fmaxf(s, 1e-24f));
        for (int d = 0; d < 32; ++d) kh[row * 32 + d] = (_Float16)(kg[row * 32 + d] * inv);
    }
    for (int tc = 0; tc < 64; ++tc)                    // lane <-> dim
        vh[lane * 64 + tc] = (_Float16)vg[lane * 64 + tc];
    __syncthreads();

    float tscale = 1.0f / fmaxf(tau[head], 0.01f);
    const float* bh = bias + (size_t)head * 4096;
    bool udm = (win >> 3) == 7;                        // bottom window row
    bool lrm = (win & 7) == 7;                         // right window column
    int mh = ((lane >> 4) & 1) * 8;

    #pragma unroll
    for (int mi = 0; mi < 4; ++mi) {                   // dots = Qn @ Kn^T
        int rowA = mi * 16 + (lane & 15);
        v16h a = frag_a(&qh[rowA * 32], lane);
        #pragma unroll
        for (int nj = 0; nj < 4; ++nj) {
            v16h bf = frag_b(&kh[(nj * 16 + (lane & 15)) * 32], lane);
            v8f c = {};
            c = wmma_f16(a, bf, c);
            int j = nj * 16 + (lane & 15);
            #pragma unroll
            for (int r = 0; r < 8; ++r) {
                int i = mi * 16 + r + mh;
                float v = c[r] * tscale + bh[i * 64 + j];
                if (udm && (((i >= 32) ? 1 : 0) != ((j >= 32) ? 1 : 0))) v += -1e9f;
                if (lrm && ((((i & 7) >= 4) ? 1 : 0) != (((j & 7) >= 4) ? 1 : 0))) v += -1e9f;
                sc[i * 64 + j] = v;
            }
        }
    }
    __syncthreads();

    #pragma unroll
    for (int rr = 0; rr < 2; ++rr) {                   // softmax, 2 rows/lane
        int row = lane * 2 + rr;
        float m = -3.4e38f;
        for (int j = 0; j < 64; ++j) m = fmaxf(m, sc[row * 64 + j]);
        float s = 0.f;
        for (int j = 0; j < 64; ++j) {
            float e = __expf(sc[row * 64 + j] - m);
            sc[row * 64 + j] = e; s += e;
        }
        float inv = 1.0f / s;
        for (int j = 0; j < 64; ++j) ah[row * 64 + j] = (_Float16)(sc[row * 64 + j] * inv);
    }
    __syncthreads();

    int wy = win >> 3, wx = win & 7;
    #pragma unroll
    for (int mi = 0; mi < 4; ++mi) {                   // out = attn @ V
        v8f oacc[2] = {};
        int rowA = mi * 16 + (lane & 15);
        #pragma unroll
        for (int kc = 0; kc < 2; ++kc) {
            v16h a = frag_a(&ah[rowA * 64 + kc * 32], lane);
            #pragma unroll
            for (int nj = 0; nj < 2; ++nj) {
                v16h bf = frag_b(&vh[(nj * 16 + (lane & 15)) * 64 + kc * 32], lane);
                oacc[nj] = wmma_f16(a, bf, oacc[nj]);
            }
        }
        #pragma unroll
        for (int nj = 0; nj < 2; ++nj) {
            int d = nj * 16 + (lane & 15);
            #pragma unroll
            for (int r = 0; r < 8; ++r) {
                int tl = mi * 16 + r + mh;
                int yy = wy * 8 + (tl >> 3), xx = wx * 8 + (tl & 7);
                ao[(((size_t)b * 64 + yy) * 64 + xx) * 256 + head * 32 + d] =
                    (_Float16)oacc[nj][r];
            }
        }
    }
}

// ---------------------------------------------------------------------------
// Kernel F: output GEMM (131072 x 256 x 256) + b_out, TDM-staged f16 tiles,
// inverse cyclic shift applied in the store index, writes d_out (b,C,64,64).
// ---------------------------------------------------------------------------
__global__ __launch_bounds__(128) void wa_out_gemm(
        const _Float16* __restrict__ ao, const _Float16* __restrict__ woT,
        const float* __restrict__ bout, float* __restrict__ out) {
    __shared__ __align__(16) _Float16 aT[64 * 32];
    __shared__ __align__(16) _Float16 bT[64 * 32];
    int mt = blockIdx.x >> 2, nt = blockIdx.x & 3;
    int b = mt >> 6, yrow = mt & 63;
    int t = threadIdx.x, lane = t & 31, wid = t >> 5;
    const _Float16* aSrc = ao + ((size_t)mt * 64 + wid * 16) * 256;
    const _Float16* bSrc = woT + ((size_t)nt * 64 + wid * 16) * 256;
    unsigned aOff = lds_off32(&aT[wid * 16 * 32]);
    unsigned bOff = lds_off32(&bT[wid * 16 * 32]);
    v8f acc[4] = {};
    for (int kk = 0; kk < 8; ++kk) {
        tdm_load_2d_f16(aSrc + kk * 32, aOff, 32, 16, 256);
        tdm_load_2d_f16(bSrc + kk * 32, bOff, 32, 16, 256);
        __builtin_amdgcn_s_wait_tensorcnt(0);
        __syncthreads();
        int row = wid * 16 + (lane & 15);
        v16h a = frag_a(&aT[row * 32], lane);
        #pragma unroll
        for (int nj = 0; nj < 4; ++nj) {
            v16h bf = frag_b(&bT[(nj * 16 + (lane & 15)) * 32], lane);
            acc[nj] = wmma_f16(a, bf, acc[nj]);
        }
        __syncthreads();
    }
    int ydst = (yrow + 4) & 63;
    int mh = ((lane >> 4) & 1) * 8;
    #pragma unroll
    for (int nj = 0; nj < 4; ++nj) {
        int c = nt * 64 + nj * 16 + (lane & 15);
        float bb = bout[c];
        #pragma unroll
        for (int r = 0; r < 8; ++r) {
            int px = wid * 16 + r + mh;
            int xdst = (px + 4) & 63;
            out[(((size_t)b * 256 + c) * 64 + ydst) * 64 + xdst] = acc[nj][r] + bb;
        }
    }
}

// ---------------------------------------------------------------------------
extern "C" void kernel_launch(void* const* d_in, const int* in_sizes, int n_in,
                              void* d_out, int out_size, void* d_ws, size_t ws_size,
                              hipStream_t stream) {
    const float* x    = (const float*)d_in[0];
    const float* tau  = (const float*)d_in[1];
    const float* wqkv = (const float*)d_in[2];
    const float* wout = (const float*)d_in[3];
    const float* bout = (const float*)d_in[4];
    const float* cw1  = (const float*)d_in[5];
    const float* cb1  = (const float*)d_in[6];
    const float* cw2  = (const float*)d_in[7];
    float* out = (float*)d_out;

    // workspace carving: q|k|vt (3 x 128MB f32), ao16 + xh16 (64MB f16 each),
    // transposed f16 weights, bias table
    const size_t QS = (size_t)32 * 8 * 64 * 64 * 32;   // 33,554,432 floats
    float* q  = (float*)d_ws;
    float* k  = q + QS;
    float* vt = k + QS;
    _Float16* ao16 = (_Float16*)(vt + QS);
    _Float16* xh16 = ao16 + (size_t)131072 * 256;
    _Float16* wqT  = xh16 + (size_t)131072 * 256;
    _Float16* woT  = wqT + 768 * 256;
    float* bias = (float*)(woT + 256 * 256);           // 8*64*64 floats

    wa_cvt_kernel<<<768, 256, 0, stream>>>(wqkv, wout, wqT, woT);
    wa_cpb_kernel<<<64, 64, 0, stream>>>(cw1, cb1, cw2, bias);
    wa_shift_kernel<<<2048, 256, 0, stream>>>(x, xh16);
    wa_qkv_gemm<<<2048 * 12, 128, 0, stream>>>(xh16, wqT, q, k, vt);
    wa_attn_kernel<<<16384, 32, 0, stream>>>(q, k, vt, tau, bias, ao16);
    wa_out_gemm<<<2048 * 4, 128, 0, stream>>>(ao16, woT, bout, out);
}
